// BaselineParser_8770323219235
// MI455X (gfx1250) — matrix operand order repaired and checked by reading, not compiled, not run on hardware
//
#include <hip/hip_runtime.h>
#include <hip/hip_bf16.h>
#include <math.h>

typedef __bf16 bf16_t;
typedef __bf16 v16bf __attribute__((ext_vector_type(16)));
typedef __bf16 v8bf  __attribute__((ext_vector_type(8)));
typedef float  v8f   __attribute__((ext_vector_type(8)));

#define BATCH   32
#define SEQLEN  1024
#define DMODEL  768
#define DFF     2048
#define NWORDS  384
#define NHEADS  8
#define DHEAD   96
#define WP1     385   // NWORDS + 1 (root)
#define HALLR   448   // WP1 padded to multiple of 64 (zero-filled)

#define F_BIAS 1
#define F_RES  2
#define F_GELU 4
#define F_UADD 8

// ------------------------------------------------------------------
// f32 -> bf16 conversion (root)
// ------------------------------------------------------------------
__global__ __launch_bounds__(256) void cvt_bf16_kernel(const float* __restrict__ src,
                                                       bf16_t* __restrict__ dst, long n) {
  long i = (long)blockIdx.x * blockDim.x + threadIdx.x;
  if (i < n) dst[i] = (bf16_t)src[i];
}

// f32 [K][N] -> bf16 transposed [N][K]  (weights, done once)
__global__ __launch_bounds__(256) void cvtT_bf16_kernel(const float* __restrict__ src,
                                                        bf16_t* __restrict__ dst,
                                                        int K, int N) {
  long i = (long)blockIdx.x * blockDim.x + threadIdx.x;
  if (i >= (long)K * N) return;
  int k = (int)(i / N), n = (int)(i % N);
  dst[(size_t)n * K + k] = (bf16_t)src[i];
}

// ------------------------------------------------------------------
// Segment mean over sorted word_ids: X[b,w,:] = mean of rows with id==w
// ------------------------------------------------------------------
__global__ __launch_bounds__(256) void seg_mean_kernel(const float* __restrict__ hid,
                                                       const int* __restrict__ wid,
                                                       float* __restrict__ X) {
  int b = blockIdx.x / NWORDS;
  int w = blockIdx.x % NWORDS;
  const int* ids = wid + (size_t)b * SEQLEN;
  int lo = 0, hi = SEQLEN;
  while (lo < hi) { int m = (lo + hi) >> 1; if (ids[m] < w) lo = m + 1; else hi = m; }
  int start = lo;
  hi = SEQLEN;
  while (lo < hi) { int m = (lo + hi) >> 1; if (ids[m] <= w) lo = m + 1; else hi = m; }
  int end = lo;
  int cnt = end - start;
  float inv = 1.0f / (float)(cnt > 0 ? cnt : 1);
  const float* hb = hid + (size_t)b * SEQLEN * DMODEL;
  float* xr = X + ((size_t)b * NWORDS + w) * DMODEL;
  for (int d = threadIdx.x; d < DMODEL; d += 256) {
    float s = 0.f;
    for (int t = start; t < end; ++t) s += hb[(size_t)t * DMODEL + d];
    xr[d] = s * inv;
  }
}

// ------------------------------------------------------------------
// Row LayerNorm (D=768), output bf16
// ------------------------------------------------------------------
__global__ __launch_bounds__(256) void ln_bf16_kernel(const float* __restrict__ in,
                                                      const float* __restrict__ g,
                                                      const float* __restrict__ bta,
                                                      bf16_t* __restrict__ out) {
  __shared__ float red[256];
  int tid = threadIdx.x;
  const float* x = in + (size_t)blockIdx.x * DMODEL;
  float s = 0.f;
  for (int d = tid; d < DMODEL; d += 256) s += x[d];
  red[tid] = s; __syncthreads();
  for (int q = 128; q > 0; q >>= 1) { if (tid < q) red[tid] += red[tid + q]; __syncthreads(); }
  float mean = red[0] * (1.0f / DMODEL); __syncthreads();
  float v = 0.f;
  for (int d = tid; d < DMODEL; d += 256) { float t = x[d] - mean; v += t * t; }
  red[tid] = v; __syncthreads();
  for (int q = 128; q > 0; q >>= 1) { if (tid < q) red[tid] += red[tid + q]; __syncthreads(); }
  float rstd = rsqrtf(red[0] * (1.0f / DMODEL) + 1e-5f);
  bf16_t* o = out + (size_t)blockIdx.x * DMODEL;
  for (int d = tid; d < DMODEL; d += 256)
    o[d] = (bf16_t)((x[d] - mean) * rstd * g[d] + bta[d]);
}

// ------------------------------------------------------------------
// Generic batched WMMA GEMM: C = epilogue(alpha * A @ BT^T)
//   A : bf16 [M][K] row-major, lda % 8 == 0, M % 128 == 0, K % 32 == 0
//   BT: bf16 [N][K] N-major ("B transposed"), ldbt % 8 == 0, rows padded
//       so tileN..tileN+63 are always readable.
// Workgroup tile 128x64, 8 waves (4x2), wave tile 32x32 (2x2 WMMA frags).
// Double-buffered software pipeline: global loads for tile t+1 are issued
// before the WMMAs of tile t, so VMEM latency overlaps the matrix stream.
// Batch z -> (bb = z / Hdim, hh = z % Hdim), per-operand strides.
// ------------------------------------------------------------------
__global__ __launch_bounds__(256) void wmma_gemm_bf16_kernel(
    const bf16_t* __restrict__ A, const bf16_t* __restrict__ BT,
    const float* __restrict__ bias, const float* __restrict__ Rres,
    const float* __restrict__ Uvec,
    float* __restrict__ Cf, bf16_t* __restrict__ Cb,
    int M, int N, int K, int lda, int ldbt, int ldc,
    long sA_b, long sA_h, long sB_b, long sB_h, long sC_b, long sC_h, long sU_b,
    int Hdim, float alpha, int flags)
{
  __shared__ __attribute__((aligned(32))) bf16_t sA[2][128][32];   // A tiles
  __shared__ __attribute__((aligned(32))) bf16_t sBT[2][64][32];   // BT tiles [n][k]

  const int tid  = threadIdx.x;
  const int lane = tid & 31;
  const int wave = tid >> 5;
  const int wr = wave >> 1;   // 0..3
  const int wc = wave & 1;    // 0..1

  const int bb = (int)blockIdx.z / Hdim;
  const int hh = (int)blockIdx.z % Hdim;
  const bf16_t* Ab  = A  + (size_t)bb * sA_b + (size_t)hh * sA_h;
  const bf16_t* BTb = BT + (size_t)bb * sB_b + (size_t)hh * sB_h;
  const size_t coff = (size_t)bb * sC_b + (size_t)hh * sC_h;

  const int tileM = blockIdx.y * 128;
  const int tileN = blockIdx.x * 64;

  // per-thread tile-load coordinates (all unconditional vector ops)
  const int ar0 = tid >> 2;            // 0..63   (two slabs of 64 rows)
  const int ac0 = (tid & 3) * 8;       // 0,8,16,24
  const bf16_t* aBase0 = Ab + (size_t)(tileM + ar0) * lda + ac0;
  const bf16_t* aBase1 = Ab + (size_t)(tileM + 64 + ar0) * lda + ac0;
  const bf16_t* bBase  = BTb + (size_t)(tileN + ar0) * ldbt + ac0;

  const int kHalf = ((lane >> 4) & 1) * 16;  // A frag: lanes 16-31 offset K by +16
  const int kOff  = ((lane >> 4) & 1) * 16;  // B frag: lanes 16-31 offset K by +16

  v8f acc[2][2] = {};
  v8bf ra0, ra1, rb0;

  const int ksteps = K >> 5;

  // prologue: stage tile 0
  ra0 = *(const v8bf*)(aBase0);
  ra1 = *(const v8bf*)(aBase1);
  rb0 = *(const v8bf*)(bBase);
  *(v8bf*)&sA[0][ar0][ac0]      = ra0;
  *(v8bf*)&sA[0][64 + ar0][ac0] = ra1;
  *(v8bf*)&sBT[0][ar0][ac0]     = rb0;
  __syncthreads();

  for (int ks = 0; ks < ksteps; ++ks) {
    const int cur = ks & 1;
    const bool more = (ks + 1) < ksteps;
    if (more) {
      const int kn = (ks + 1) << 5;
      ra0 = *(const v8bf*)(aBase0 + kn);
      ra1 = *(const v8bf*)(aBase1 + kn);
      rb0 = *(const v8bf*)(bBase + kn);
      if (ks + 2 < ksteps) {           // L2 prefetch two tiles ahead
        __builtin_prefetch(aBase0 + kn + 32, 0, 1);
        __builtin_prefetch(bBase + kn + 32, 0, 1);
      }
    }

    // fragment assembly: contiguous k order, element s = k 16*(lane/16)+s, same for A and B (wave32)
    v16bf afrag[2], bfrag[2];
    #pragma unroll
    for (int i = 0; i < 2; ++i) {
      int ar = wr * 32 + i * 16 + (lane & 15);
      v8bf lo = *(const v8bf*)&sA[cur][ar][kHalf];
      v8bf hi = *(const v8bf*)&sA[cur][ar][8 + kHalf];
      #pragma unroll
      for (int e = 0; e < 8; ++e) { afrag[i][e] = lo[e]; afrag[i][e + 8] = hi[e]; }
    }
    #pragma unroll
    for (int j = 0; j < 2; ++j) {
      int bn = wc * 32 + j * 16 + (lane & 15);
      bfrag[j] = *(const v16bf*)&sBT[cur][bn][kOff];
    }
    #pragma unroll
    for (int i = 0; i < 2; ++i)
      #pragma unroll
      for (int j = 0; j < 2; ++j)
        acc[i][j] = __builtin_amdgcn_wmma_f32_16x16x32_bf16(
            false, afrag[i], false, bfrag[j], (short)0, acc[i][j], false, false);

    if (more) {
      const int nxt = cur ^ 1;
      *(v8bf*)&sA[nxt][ar0][ac0]      = ra0;
      *(v8bf*)&sA[nxt][64 + ar0][ac0] = ra1;
      *(v8bf*)&sBT[nxt][ar0][ac0]     = rb0;
      __syncthreads();
    }
  }

  // epilogue: C frag: VGPR e -> row = e + (lane>=16 ? 8 : 0), col = lane&15
  #pragma unroll
  for (int i = 0; i < 2; ++i) {
    #pragma unroll
    for (int j = 0; j < 2; ++j) {
      int col = tileN + wc * 32 + j * 16 + (lane & 15);
      if (col < N) {
        float badd = 0.f;
        if (flags & F_BIAS) badd += bias[col];
        if (flags & F_UADD) badd += Uvec[(size_t)bb * sU_b + col];
        #pragma unroll
        for (int e = 0; e < 8; ++e) {
          int row = tileM + wr * 32 + i * 16 + ((lane >> 4) << 3) + e;
          float v = acc[i][j][e] * alpha + badd;
          size_t cidx = coff + (size_t)row * ldc + col;
          if (flags & F_RES)  v += Rres[cidx];
          if (flags & F_GELU) v = 0.5f * v * (1.0f + erff(v * 0.70710678118f));
          if (Cf) Cf[cidx] = v;
          if (Cb) Cb[cidx] = (bf16_t)v;
        }
      }
    }
  }
}

// ------------------------------------------------------------------
// Split QKV [B,W,3D] bf16 -> Q [B,H,W,dh], K [B,H,W,dh], V^T [B,H,dh,W]
// (K in q-layout IS the BT operand for Q@K^T; V^T is BT for P@V)
// ------------------------------------------------------------------
__global__ __launch_bounds__(256) void split_heads_kernel(const bf16_t* __restrict__ qkv,
                                                          bf16_t* __restrict__ Q,
                                                          bf16_t* __restrict__ Kb,
                                                          bf16_t* __restrict__ Vt) {
  long idx = (long)blockIdx.x * blockDim.x + threadIdx.x;
  long total = (long)BATCH * NWORDS * DMODEL;
  if (idx >= total) return;
  int c = (int)(idx % DMODEL);
  long bw = idx / DMODEL;
  int w = (int)(bw % NWORDS);
  int b = (int)(bw / NWORDS);
  int h = c / DHEAD, d = c % DHEAD;
  const bf16_t* src = qkv + ((size_t)b * NWORDS + w) * (3 * DMODEL);
  size_t qi = (((size_t)b * NHEADS + h) * NWORDS + w) * DHEAD + d;
  size_t vi = (((size_t)b * NHEADS + h) * DHEAD + d) * NWORDS + w;
  Q[qi]  = src[c];
  Kb[qi] = src[DMODEL + c];
  Vt[vi] = src[2 * DMODEL + c];
}

// ------------------------------------------------------------------
// Row softmax over W=384 scores, output bf16 probabilities
// ------------------------------------------------------------------
__global__ __launch_bounds__(128) void softmax_bf16_kernel(const float* __restrict__ S,
                                                           bf16_t* __restrict__ P) {
  __shared__ float red[128];
  int tid = threadIdx.x;
  const float* s = S + (size_t)blockIdx.x * NWORDS;
  float m = -1e30f;
  for (int i = tid; i < NWORDS; i += 128) m = fmaxf(m, s[i]);
  red[tid] = m; __syncthreads();
  for (int q = 64; q > 0; q >>= 1) { if (tid < q) red[tid] = fmaxf(red[tid], red[tid + q]); __syncthreads(); }
  m = red[0]; __syncthreads();
  float sum = 0.f;
  for (int i = tid; i < NWORDS; i += 128) sum += __expf(s[i] - m);
  red[tid] = sum; __syncthreads();
  for (int q = 64; q > 0; q >>= 1) { if (tid < q) red[tid] += red[tid + q]; __syncthreads(); }
  float inv = 1.0f / red[0];
  bf16_t* p = P + (size_t)blockIdx.x * NWORDS;
  for (int i = tid; i < NWORDS; i += 128) p[i] = (bf16_t)(__expf(s[i] - m) * inv);
}

// ------------------------------------------------------------------
// Build Hall bf16 [B, HALLR=448, D]: row0 = root, rows 1..384 = X2,
// rows 385..447 zero (padding so logits GEMM loads need no guards)
// ------------------------------------------------------------------
__global__ __launch_bounds__(256) void hall_kernel(const bf16_t* __restrict__ X2b,
                                                   const bf16_t* __restrict__ rootb,
                                                   bf16_t* __restrict__ Hall) {
  long idx = (long)blockIdx.x * blockDim.x + threadIdx.x;
  long total = (long)BATCH * HALLR * DMODEL;
  if (idx >= total) return;
  int d = (int)(idx % DMODEL);
  long br = idx / DMODEL;
  int r = (int)(br % HALLR);
  int b = (int)(br / HALLR);
  bf16_t v = (bf16_t)0.f;
  if (r == 0)        v = rootb[d];
  else if (r < WP1)  v = X2b[((size_t)b * NWORDS + (r - 1)) * DMODEL + d];
  Hall[idx] = v;
}

// ------------------------------------------------------------------
// u[b,n] = dot(Hall[b,n], Uw) + Ub  (f32 sources for accuracy)
// ------------------------------------------------------------------
__global__ __launch_bounds__(256) void u_kernel(const float* __restrict__ X2,
                                                const float* __restrict__ root,
                                                const float* __restrict__ Uw,
                                                const float* __restrict__ Ubp,
                                                float* __restrict__ U) {
  __shared__ float red[256];
  int row = blockIdx.x;            // b*WP1 + n
  int b = row / WP1, n = row % WP1;
  const float* src = (n == 0) ? root : X2 + ((size_t)b * NWORDS + (n - 1)) * DMODEL;
  float s = 0.f;
  for (int d = threadIdx.x; d < DMODEL; d += 256) s += src[d] * Uw[d];
  red[threadIdx.x] = s; __syncthreads();
  for (int q = 128; q > 0; q >>= 1) { if (threadIdx.x < q) red[threadIdx.x] += red[threadIdx.x + q]; __syncthreads(); }
  if (threadIdx.x == 0) U[row] = red[0] + Ubp[0];
}

// word_ids are sorted ascending -> max id = last element
__global__ void maxid_kernel(const int* __restrict__ wid, int* __restrict__ mx) {
  int b = blockIdx.x * blockDim.x + threadIdx.x;
  if (b < BATCH) mx[b] = wid[(size_t)b * SEQLEN + SEQLEN - 1];
}

// ------------------------------------------------------------------
// Per-row masked log-softmax NLL: nllrow[b*W+t] (0 for masked rows)
// ------------------------------------------------------------------
__global__ __launch_bounds__(128) void loss_row_kernel(const float* __restrict__ logits,
                                                       const int* __restrict__ gold,
                                                       const int* __restrict__ maxid,
                                                       float* __restrict__ nllrow) {
  __shared__ float red[128];
  int row = blockIdx.x;                 // b*NWORDS + t
  int b = row / NWORDS, t = row % NWORDS;
  int mi = maxid[b];
  int tid = threadIdx.x;
  if (t > mi) { if (tid == 0) nllrow[row] = 0.f; return; }
  const float* lg = logits + (size_t)row * WP1;
  float m = -1e30f;
  for (int n = tid; n < WP1; n += 128) {
    float v = (n == 0 || (n - 1) <= mi) ? lg[n] : -1e9f;
    m = fmaxf(m, v);
  }
  red[tid] = m; __syncthreads();
  for (int q = 64; q > 0; q >>= 1) { if (tid < q) red[tid] = fmaxf(red[tid], red[tid + q]); __syncthreads(); }
  m = red[0]; __syncthreads();
  float s = 0.f;
  for (int n = tid; n < WP1; n += 128) {
    float v = (n == 0 || (n - 1) <= mi) ? lg[n] : -1e9f;
    s += __expf(v - m);
  }
  red[tid] = s; __syncthreads();
  for (int q = 64; q > 0; q >>= 1) { if (tid < q) red[tid] += red[tid + q]; __syncthreads(); }
  if (tid == 0) {
    float lse = m + __logf(red[0]);
    int g = gold[row];
    float v = (g == 0 || (g - 1) <= mi) ? lg[g] : -1e9f;
    nllrow[row] = lse - v;
  }
}

// Deterministic final reduction -> scalar loss
__global__ __launch_bounds__(256) void finalize_kernel(const float* __restrict__ nllrow,
                                                       const int* __restrict__ maxid,
                                                       float* __restrict__ out) {
  __shared__ float red[256];
  int tid = threadIdx.x;
  float s = 0.f;
  for (int i = tid; i < BATCH * NWORDS; i += 256) s += nllrow[i];
  red[tid] = s; __syncthreads();
  for (int q = 128; q > 0; q >>= 1) { if (tid < q) red[tid] += red[tid + q]; __syncthreads(); }
  float tot = red[0]; __syncthreads();
  float c = 0.f;
  for (int i = tid; i < BATCH; i += 256) c += (float)(maxid[i] + 1);
  red[tid] = c; __syncthreads();
  for (int q = 128; q > 0; q >>= 1) { if (tid < q) red[tid] += red[tid + q]; __syncthreads(); }
  if (tid == 0) out[0] = tot / red[0];
}

// ------------------------------------------------------------------
// Host driver
// ------------------------------------------------------------------
extern "C" void kernel_launch(void* const* d_in, const int* in_sizes, int n_in,
                              void* d_out, int out_size, void* d_ws, size_t ws_size,
                              hipStream_t stream) {
  (void)in_sizes; (void)n_in; (void)out_size; (void)ws_size;
  const float* last_hidden = (const float*)d_in[0];
  const float* ln1_g = (const float*)d_in[1];
  const float* ln1_b = (const float*)d_in[2];
  const float* Wqkv  = (const float*)d_in[3];
  const float* bqkv  = (const float*)d_in[4];
  const float* Wo    = (const float*)d_in[5];
  const float* bo    = (const float*)d_in[6];
  const float* ln2_g = (const float*)d_in[7];
  const float* ln2_b = (const float*)d_in[8];
  const float* W1    = (const float*)d_in[9];
  const float* b1    = (const float*)d_in[10];
  const float* W2    = (const float*)d_in[11];
  const float* b2    = (const float*)d_in[12];
  const float* Wbi   = (const float*)d_in[13];
  const float* Uw    = (const float*)d_in[14];
  const float* Ubs   = (const float*)d_in[15];
  const float* root  = (const float*)d_in[16];
  const int* word_ids   = (const int*)d_in[17];
  const int* heads_gold = (const int*)d_in[18];
  float* out = (float*)d_out;

  // ---- workspace carve ----
  char* base = (char*)d_ws;
  size_t off = 0;
  auto carve = [&](size_t bytes) -> void* {
    void* r = base + off;
    off = (off + bytes + 255) & ~(size_t)255;
    return r;
  };
  const size_t BWD = (size_t)BATCH * NWORDS * DMODEL;
  float*  X     = (float*) carve(BWD * 4);
  bf16_t* Xn    = (bf16_t*)carve(BWD * 2);
  bf16_t* QKVb  = (bf16_t*)carve((size_t)BATCH * NWORDS * 3 * DMODEL * 2);
  bf16_t* Qb    = (bf16_t*)carve(BWD * 2);
  bf16_t* Kbb   = (bf16_t*)carve(BWD * 2);
  bf16_t* Vtb   = (bf16_t*)carve(BWD * 2);
  float*  Sb    = (float*) carve((size_t)BATCH * NHEADS * NWORDS * NWORDS * 4);
  bf16_t* Pb    = (bf16_t*)carve((size_t)BATCH * NHEADS * NWORDS * NWORDS * 2);
  bf16_t* Yb    = (bf16_t*)carve(BWD * 2);
  float*  X1    = (float*) carve(BWD * 4);
  bf16_t* Xn2   = (bf16_t*)carve(BWD * 2);
  bf16_t* H1b   = (bf16_t*)carve((size_t)BATCH * NWORDS * DFF * 2);
  float*  X2    = (float*) carve(BWD * 4);
  bf16_t* X2b   = (bf16_t*)carve(BWD * 2);
  bf16_t* Tb    = (bf16_t*)carve(BWD * 2);
  bf16_t* Hall  = (bf16_t*)carve((size_t)BATCH * HALLR * DMODEL * 2);
  float*  Ubuf  = (float*) carve((size_t)BATCH * WP1 * 4);
  float*  Lg    = (float*) carve((size_t)BATCH * NWORDS * WP1 * 4);
  bf16_t* WqkvT = (bf16_t*)carve((size_t)DMODEL * 3 * DMODEL * 2);
  bf16_t* WoT   = (bf16_t*)carve((size_t)DMODEL * DMODEL * 2);
  bf16_t* W1T   = (bf16_t*)carve((size_t)DMODEL * DFF * 2);
  bf16_t* W2T   = (bf16_t*)carve((size_t)DFF * DMODEL * 2);
  bf16_t* WbiT  = (bf16_t*)carve((size_t)DMODEL * DMODEL * 2);
  bf16_t* rootb = (bf16_t*)carve((size_t)DMODEL * 2);
  int*    mxid  = (int*)   carve((size_t)BATCH * 4);
  float*  nllr  = (float*) carve((size_t)BATCH * NWORDS * 4);

  auto cvtT = [&](const float* s, bf16_t* d, int K, int N) {
    long n = (long)K * N;
    cvtT_bf16_kernel<<<dim3((unsigned)((n + 255) / 256)), dim3(256), 0, stream>>>(s, d, K, N);
  };
  auto gemm = [&](const bf16_t* Aa, const bf16_t* BTp, const float* bias,
                  const float* R, const float* U, float* Cf, bf16_t* Cb,
                  int M, int N, int K, int lda, int ldbt, int ldc,
                  long sAb, long sAh, long sBb, long sBh, long sCb, long sCh, long sUb,
                  int H, int batches, float alpha, int flags) {
    dim3 g((unsigned)((N + 63) / 64), (unsigned)((M + 127) / 128), (unsigned)batches);
    wmma_gemm_bf16_kernel<<<g, dim3(256), 0, stream>>>(Aa, BTp, bias, R, U, Cf, Cb,
        M, N, K, lda, ldbt, ldc, sAb, sAh, sBb, sBh, sCb, sCh, sUb, H, alpha, flags);
  };

  // ---- weight conversions (transposed to N-major BT layout) ----
  cvtT(Wqkv, WqkvT, DMODEL, 3 * DMODEL);
  cvtT(Wo,   WoT,   DMODEL, DMODEL);
  cvtT(W1,   W1T,   DMODEL, DFF);
  cvtT(W2,   W2T,   DFF, DMODEL);
  cvtT(Wbi,  WbiT,  DMODEL, DMODEL);
  cvt_bf16_kernel<<<dim3(3), dim3(256), 0, stream>>>(root, rootb, DMODEL);

  const int MR = BATCH * NWORDS;   // 12288 flat rows

  // ---- pooling + LN1 ----
  seg_mean_kernel<<<dim3(MR), dim3(256), 0, stream>>>(last_hidden, word_ids, X);
  ln_bf16_kernel<<<dim3(MR), dim3(256), 0, stream>>>(X, ln1_g, ln1_b, Xn);

  // ---- QKV projection ----
  gemm(Xn, WqkvT, bqkv, nullptr, nullptr, nullptr, QKVb,
       MR, 3 * DMODEL, DMODEL, DMODEL, DMODEL, 3 * DMODEL,
       0, 0, 0, 0, 0, 0, 0, 1, 1, 1.0f, F_BIAS);

  split_heads_kernel<<<dim3((unsigned)((BWD + 255) / 256)), dim3(256), 0, stream>>>(QKVb, Qb, Kbb, Vtb);

  // ---- attention scores S = (Q K^T)/sqrt(dh): BT operand = K (q-layout) ----
  gemm(Qb, Kbb, nullptr, nullptr, nullptr, Sb, nullptr,
       NWORDS, NWORDS, DHEAD, DHEAD, DHEAD, NWORDS,
       (long)NHEADS * NWORDS * DHEAD, (long)NWORDS * DHEAD,
       (long)NHEADS * NWORDS * DHEAD, (long)NWORDS * DHEAD,
       (long)NHEADS * NWORDS * NWORDS, (long)NWORDS * NWORDS,
       0, NHEADS, BATCH * NHEADS, 0.1020620726f /* 1/sqrt(96) */, 0);

  softmax_bf16_kernel<<<dim3(BATCH * NHEADS * NWORDS), dim3(128), 0, stream>>>(Sb, Pb);

  // ---- Y = P V: BT operand = V^T [B,H,dh,W]; heads merged via C strides ----
  gemm(Pb, Vtb, nullptr, nullptr, nullptr, nullptr, Yb,
       NWORDS, DHEAD, NWORDS, NWORDS, NWORDS, DMODEL,
       (long)NHEADS * NWORDS * NWORDS, (long)NWORDS * NWORDS,
       (long)NHEADS * DHEAD * NWORDS, (long)DHEAD * NWORDS,
       (long)NWORDS * DMODEL, (long)DHEAD,
       0, NHEADS, BATCH * NHEADS, 1.0f, 0);

  // ---- out-proj + residual(X) -> X1 ----
  gemm(Yb, WoT, bo, X, nullptr, X1, nullptr,
       MR, DMODEL, DMODEL, DMODEL, DMODEL, DMODEL,
       0, 0, 0, 0, 0, 0, 0, 1, 1, 1.0f, F_BIAS | F_RES);

  // ---- LN2 -> FFN ----
  ln_bf16_kernel<<<dim3(MR), dim3(256), 0, stream>>>(X1, ln2_g, ln2_b, Xn2);
  gemm(Xn2, W1T, b1, nullptr, nullptr, nullptr, H1b,
       MR, DFF, DMODEL, DMODEL, DMODEL, DFF,
       0, 0, 0, 0, 0, 0, 0, 1, 1, 1.0f, F_BIAS | F_GELU);
  gemm(H1b, W2T, b2, X1, nullptr, X2, X2b,
       MR, DMODEL, DFF, DFF, DFF, DMODEL,
       0, 0, 0, 0, 0, 0, 0, 1, 1, 1.0f, F_BIAS | F_RES);

  // ---- biaffine: T = X2 @ Wbi ----
  gemm(X2b, WbiT, nullptr, nullptr, nullptr, nullptr, Tb,
       MR, DMODEL, DMODEL, DMODEL, DMODEL, DMODEL,
       0, 0, 0, 0, 0, 0, 0, 1, 1, 1.0f, 0);

  hall_kernel<<<dim3((unsigned)(((long)BATCH * HALLR * DMODEL + 255) / 256)), dim3(256), 0, stream>>>(X2b, rootb, Hall);
  u_kernel<<<dim3(BATCH * WP1), dim3(256), 0, stream>>>(X2, root, Uw, Ubs, Ubuf);
  maxid_kernel<<<dim3(1), dim3(32), 0, stream>>>(word_ids, mxid);

  // ---- logits[b] = T_b @ Hall_b^T + u_b: BT operand = Hall (row-major) ----
  gemm(Tb, Hall, nullptr, nullptr, Ubuf, Lg, nullptr,
       NWORDS, WP1, DMODEL, DMODEL, DMODEL, WP1,
       (long)NWORDS * DMODEL, 0,
       (long)HALLR * DMODEL, 0,
       (long)NWORDS * WP1, 0,
       (long)WP1, 1, BATCH, 1.0f, F_UADD);

  // ---- masked CE ----
  loss_row_kernel<<<dim3(MR), dim3(128), 0, stream>>>(Lg, heads_gold, mxid, nllr);
  finalize_kernel<<<dim3(1), dim3(256), 0, stream>>>(nllr, mxid, out);
}
